// IsoNeighborNN_57939108823864
// MI455X (gfx1250) — matrix operand order repaired and checked by reading, not compiled, not run on hardware
//
#include <hip/hip_runtime.h>
#include <hip/hip_bf16.h>
#include <stdint.h>

#define HID   256
#define NNB   32
#define NATM  2048
#define NBAT  4
#define WAVES 8
#define ROWS_PER_BLOCK 128   // 8 waves x 16 pairs = 4 atoms x 32 neighbors
#define BOXL  10.0f

#define SLAB_ELE (HID * 32)              // 8192 f16 elements = 16 KB per k-slab
#define OFF_W0F  0                        // layer0 fwd (K padded 5->32), 1 slab
#define OFF_W1F  (OFF_W0F + SLAB_ELE)     // W1 fwd, 8 slabs
#define OFF_W2F  (OFF_W1F + 8 * SLAB_ELE) // W2 fwd, 8 slabs
#define OFF_W2B  (OFF_W2F + 8 * SLAB_ELE) // W2^T bwd, 8 slabs
#define OFF_W1B  (OFF_W2B + 8 * SLAB_ELE) // W1^T bwd, 8 slabs
#define OFF_W0B  (OFF_W1B + 8 * SLAB_ELE) // W0^T bwd (5 valid rows), 8 slabs
#define WS_TOTAL (OFF_W0B + 8 * SLAB_ELE) // 335872 halfs = 656 KB

typedef __attribute__((ext_vector_type(16))) _Float16 v16h;
typedef __attribute__((ext_vector_type(8)))  _Float16 v8h;
typedef __attribute__((ext_vector_type(8)))  float    v8f;
typedef __attribute__((ext_vector_type(4)))  unsigned int u32x4;
typedef __attribute__((ext_vector_type(8)))  int      i32x8;
typedef __attribute__((ext_vector_type(4)))  int      i32x4;

#if defined(__HIP_DEVICE_COMPILE__) && defined(__has_builtin)
# if __has_builtin(__builtin_amdgcn_tensor_load_to_lds) && __has_builtin(__builtin_amdgcn_s_wait_tensorcnt)
#  define HAVE_TDM 1
# else
#  define HAVE_TDM 0
# endif
#else
# define HAVE_TDM 0
#endif

#if HAVE_TDM
// One 1-D TDM transfer: 8192 x 2-byte elements (16 KB) global -> LDS.
// D# per CDNA5 ISA (type=2, count=1, no multicast,
// no padding/iteration/gather; tile_dim0 = tensor_dim0 = 8192).
__device__ __forceinline__ void tdm_issue(uint32_t lds_byte, const void* gptr) {
  const uint64_t ga = (uint64_t)(uintptr_t)gptr;
  u32x4 g0;
  g0[0] = 1u;                                            // count = 1 (valid descriptor)
  g0[1] = lds_byte;                                      // LDS byte address
  g0[2] = (uint32_t)(ga & 0xFFFFFFFFu);                  // global addr [31:0]
  g0[3] = (uint32_t)((ga >> 32) & 0x01FFFFFFu) | (2u << 30); // addr[56:32] | type=2
  i32x8 g1;
  g1[0] = (int)(1u << 16);        // workgroup_mask=0, data_size=1 (2 bytes)
  g1[1] = (int)(0x2000u << 16);   // tensor_dim0[15:0] = 8192 (in [31:16])
  g1[2] = (int)(1u << 16);        // tensor_dim0[31:16]=0 ; tensor_dim1[15:0]=1
  g1[3] = (int)(0x2000u << 16);   // tensor_dim1[31:16]=0 ; tile_dim0 = 8192
  g1[4] = 0;                      // tile_dim1 = 0 (unused), tile_dim2 = 0
  g1[5] = 0x2000;                 // tensor_dim0_stride = 8192
  g1[6] = 0;
  g1[7] = 0;
  const i32x4 z4 = {0, 0, 0, 0};
#if __clang_major__ >= 23
  const i32x8 z8 = {0, 0, 0, 0, 0, 0, 0, 0};
  __builtin_amdgcn_tensor_load_to_lds(g0, g1, z4, z4, z8, 0);
#else
  __builtin_amdgcn_tensor_load_to_lds(g0, g1, z4, z4, 0);
#endif
}
#endif

// A fragment: row m = lane%16; halfs e0..15 -> K = base..base+15,
// base = k0 + (lane<16 ? 0 : 16).  Two contiguous 16B LDS loads.
__device__ __forceinline__ v16h load_a_frag(const _Float16* hrow, int k0, int sel8) {
  v8h a0 = *(const v8h*)(hrow + k0 + 2 * sel8);
  v8h a1 = *(const v8h*)(hrow + k0 + 2 * sel8 + 8);
  v16h a;
#pragma unroll
  for (int i = 0; i < 8; ++i) { a[i] = a0[i]; a[i + 8] = a1[i]; }
  return a;
}

__device__ __forceinline__ void wmma16(v8f (&acc)[16], const _Float16 (*sl)[32],
                                       v16h a, int nr, int sel16) {
#pragma unroll
  for (int t = 0; t < 16; ++t) {
    v16h b = *(const v16h*)(&sl[t * 16 + nr][0] + sel16);
    acc[t] = __builtin_amdgcn_wmma_f32_16x16x32_f16(
        false, a, false, b, (short)0, acc[t], false, false);
  }
}

// K=256 GEMM layer (8 k-steps), B-slabs pre-packed in global, streamed by TDM
// into a double-buffered 2x16KB LDS slab; WMMA overlaps the next slab's DMA.
__device__ __forceinline__ void gemm256_pk(v8f (&acc)[16], const _Float16* myArow,
                                           _Float16 (*slab2)[HID][32],
                                           const _Float16* gslabs,
                                           int tid, int wave, int nr, int sel8, int sel16) {
#if HAVE_TDM
  __syncthreads();                       // all waves done with previous slab user
  if (wave == 0) tdm_issue((uint32_t)(uintptr_t)&slab2[0][0][0], gslabs);
#pragma unroll 1
  for (int ks = 0; ks < 8; ++ks) {
    if (wave == 0) __builtin_amdgcn_s_wait_tensorcnt(0);
    __syncthreads();                     // slab[ks&1] ready for everyone
    if (wave == 0 && ks < 7)
      tdm_issue((uint32_t)(uintptr_t)&slab2[(ks + 1) & 1][0][0],
                gslabs + (ks + 1) * SLAB_ELE);
    v16h a = load_a_frag(myArow, ks * 32, sel8);
    wmma16(acc, slab2[ks & 1], a, nr, sel16);
  }
#else
#pragma unroll 1
  for (int ks = 0; ks < 8; ++ks) {
    __syncthreads();
    {
      const float4* src = (const float4*)(gslabs + ks * SLAB_ELE);
      float4* dst = (float4*)&slab2[0][0][0];
#pragma unroll
      for (int q = 0; q < 4; ++q) dst[q * 256 + tid] = src[q * 256 + tid];
    }
    __syncthreads();
    v16h a = load_a_frag(myArow, ks * 32, sel8);
    wmma16(acc, slab2[0], a, nr, sel16);
  }
#endif
}

__device__ __forceinline__ void relu_store(v8f (&acc)[16], const float* __restrict__ bg,
                                           uint32_t* mask, _Float16 (*hw)[HID],
                                           int nr, int moff) {
#pragma unroll
  for (int t = 0; t < 16; ++t) {
    const float bn = bg[t * 16 + nr];
#pragma unroll
    for (int v = 0; v < 8; ++v) {
      float x = acc[t][v] + bn;
      const bool on = x > 0.0f;
      if (on) mask[t >> 2] |= (1u << ((t * 8 + v) & 31));
      hw[v + moff][t * 16 + nr] = on ? (_Float16)x : (_Float16)0.0f;
    }
  }
}

__device__ __forceinline__ void mask_store(v8f (&acc)[16], const uint32_t* mask,
                                           _Float16 (*hw)[HID], int nr, int moff) {
#pragma unroll
  for (int t = 0; t < 16; ++t) {
#pragma unroll
    for (int v = 0; v < 8; ++v) {
      const bool on = (mask[t >> 2] >> ((t * 8 + v) & 31)) & 1u;
      hw[v + moff][t * 16 + nr] = on ? (_Float16)acc[t][v] : (_Float16)0.0f;
    }
  }
}

// ---------------- prep: convert f32 weights -> f16 slabs in exact LDS image layout ----------------
__global__ __launch_bounds__(256)
void prep_weights_kernel(const float* __restrict__ W0, const float* __restrict__ W1,
                         const float* __restrict__ W2, _Float16* __restrict__ ws) {
  const int e = blockIdx.x * 256 + threadIdx.x;
  if (e >= WS_TOTAL) return;
  float v;
  if (e < OFF_W1F) {                               // fwd L0: [n][kk] = kk<5 ? W0[kk][n] : 0
    const int n = e >> 5, kk = e & 31;
    v = (kk < 5) ? W0[kk * HID + n] : 0.0f;
  } else if (e < OFF_W2F) {                        // fwd W1: [ks][n][kk] = W1[k][n]
    const int r = e - OFF_W1F;
    const int k = (r / SLAB_ELE) * 32 + (r & 31), n = (r % SLAB_ELE) >> 5;
    v = W1[k * HID + n];
  } else if (e < OFF_W2B) {                        // fwd W2
    const int r = e - OFF_W2F;
    const int k = (r / SLAB_ELE) * 32 + (r & 31), n = (r % SLAB_ELE) >> 5;
    v = W2[k * HID + n];
  } else if (e < OFF_W1B) {                        // bwd W2^T: [ks][n][kk] = W2[n][k]
    const int r = e - OFF_W2B;
    const int k = (r / SLAB_ELE) * 32 + (r & 31), n = (r % SLAB_ELE) >> 5;
    v = W2[n * HID + k];
  } else if (e < OFF_W0B) {                        // bwd W1^T
    const int r = e - OFF_W1B;
    const int k = (r / SLAB_ELE) * 32 + (r & 31), n = (r % SLAB_ELE) >> 5;
    v = W1[n * HID + k];
  } else {                                         // bwd W0^T: [ks][n][kk] = n<5 ? W0[n][k] : 0
    const int r = e - OFF_W0B;
    const int k = (r / SLAB_ELE) * 32 + (r & 31), n = (r % SLAB_ELE) >> 5;
    v = (n < 5) ? W0[n * HID + k] : 0.0f;
  }
  ws[e] = (_Float16)v;
}

// ---------------- fused forward + input-gradient + force reduction ----------------
__global__ __launch_bounds__(256, 1)
void pairmlp_force_kernel(const float* __restrict__ pos,
                          const int*   __restrict__ nl,
                          const _Float16* __restrict__ ws,
                          const float* __restrict__ b0g, const float* __restrict__ b1g,
                          const float* __restrict__ b2g, const float* __restrict__ W3g,
                          float* __restrict__ out) {
  __shared__ _Float16 s_slab[2][HID][32];     // 32 KB  double-buffered B-operand k-slab
  __shared__ _Float16 s_h[WAVES][16][HID];    // 64 KB  per-wave activations / grads (f16)
  __shared__ float    s_gf[WAVES][16][16];    // 8 KB   per-wave gfeat staging
  __shared__ float    s_fsum[4][3];           // per-atom force accumulators

  const int tid   = threadIdx.x;
  const int wave  = tid >> 5;
  const int lane  = tid & 31;
  const int nr    = lane & 15;
  const int sel8  = (lane < 16) ? 0 : 8;
  const int sel16 = (lane < 16) ? 0 : 16;
  const int moff  = (lane < 16) ? 0 : 8;

  if (tid < 12) s_fsum[tid / 3][tid % 3] = 0.0f;

  // ---- per-pair features (f32 scalar path, lanes 0..15) ----
  float dx = 0.f, dy = 0.f, dz = 0.f, R = 1.f;
  if (lane < 16) {
    const int p    = blockIdx.x * ROWS_PER_BLOCK + wave * 16 + lane;
    const int atom = p >> 5;               // = b*N + i
    const int b    = atom / NATM;
    const int j    = nl[2 * p + 1];
    const int sb   = atom * 3;
    const int ob   = (b * NATM + j) * 3;
    dx = pos[sb + 0] - pos[ob + 0];
    dy = pos[sb + 1] - pos[ob + 1];
    dz = pos[sb + 2] - pos[ob + 2];
    const float half = 0.5f * BOXL;
    dx = (dx >  half) ? dx - BOXL : dx;  dx = (dx < -half) ? dx + BOXL : dx;
    dy = (dy >  half) ? dy - BOXL : dy;  dy = (dy < -half) ? dy + BOXL : dy;
    dz = (dz >  half) ? dz - BOXL : dz;  dz = (dz < -half) ? dz + BOXL : dz;
    R = sqrtf(dx * dx + dy * dy + dz * dz);
    const float Rinv = 1.0f / R;
    _Float16* row = &s_h[wave][lane][0];
    row[0] = (_Float16)dx; row[1] = (_Float16)dy; row[2] = (_Float16)dz;
    row[3] = (_Float16)R;  row[4] = (_Float16)Rinv;
#pragma unroll
    for (int k = 5; k < 32; ++k) row[k] = (_Float16)0.0f;
  }

  uint32_t mask0[4] = {0, 0, 0, 0};
  uint32_t mask1[4] = {0, 0, 0, 0};
  const _Float16* myArow = &s_h[wave][nr][0];
  _Float16 (*hw)[HID] = s_h[wave];

  // ---------------- Layer 0: feats(16x32pad) @ W0 ----------------
  {
    v8f acc[16] = {};
#if HAVE_TDM
    __syncthreads();
    if (wave == 0) {
      tdm_issue((uint32_t)(uintptr_t)&s_slab[0][0][0], ws + OFF_W0F);
      __builtin_amdgcn_s_wait_tensorcnt(0);
    }
    __syncthreads();
#else
    __syncthreads();
    {
      const float4* src = (const float4*)(ws + OFF_W0F);
      float4* dst = (float4*)&s_slab[0][0][0];
#pragma unroll
      for (int q = 0; q < 4; ++q) dst[q * 256 + tid] = src[q * 256 + tid];
    }
    __syncthreads();
#endif
    v16h a = load_a_frag(myArow, 0, sel8);
    wmma16(acc, s_slab[0], a, nr, sel16);
    relu_store(acc, b0g, mask0, hw, nr, moff);          // h0 -> s_h, record mask0
  }

  // ---------------- Layer 1: h0 @ W1 ----------------
  {
    v8f acc[16] = {};
    gemm256_pk(acc, myArow, s_slab, ws + OFF_W1F, tid, wave, nr, sel8, sel16);
    relu_store(acc, b1g, mask1, hw, nr, moff);          // h1 -> s_h, record mask1
  }

  // ---------------- Layer 2: h1 @ W2 ; seed g2 = mask2 * W3 ----------------
  {
    v8f acc[16] = {};
    gemm256_pk(acc, myArow, s_slab, ws + OFF_W2F, tid, wave, nr, sel8, sel16);
#pragma unroll
    for (int t = 0; t < 16; ++t) {
      const float bn  = b2g[t * 16 + nr];
      const float w3n = W3g[t * 16 + nr];
#pragma unroll
      for (int v = 0; v < 8; ++v) {
        const float x = acc[t][v] + bn;
        hw[v + moff][t * 16 + nr] = (x > 0.0f) ? (_Float16)w3n : (_Float16)0.0f;
      }
    }
  }

  // ---------------- Backward: g1 = (g2 @ W2^T) * mask1 ----------------
  {
    v8f acc[16] = {};
    gemm256_pk(acc, myArow, s_slab, ws + OFF_W2B, tid, wave, nr, sel8, sel16);
    mask_store(acc, mask1, hw, nr, moff);
  }

  // ---------------- Backward: g0 = (g1 @ W1^T) * mask0 ----------------
  {
    v8f acc[16] = {};
    gemm256_pk(acc, myArow, s_slab, ws + OFF_W1B, tid, wave, nr, sel8, sel16);
    mask_store(acc, mask0, hw, nr, moff);
  }

  // ---------------- gfeat = g0 @ W0^T (single 16-wide N tile, cols 0..4 valid) ----------------
  v8f accf = {};
#if HAVE_TDM
  __syncthreads();
  if (wave == 0) tdm_issue((uint32_t)(uintptr_t)&s_slab[0][0][0], ws + OFF_W0B);
#pragma unroll 1
  for (int ks = 0; ks < 8; ++ks) {
    if (wave == 0) __builtin_amdgcn_s_wait_tensorcnt(0);
    __syncthreads();
    if (wave == 0 && ks < 7)
      tdm_issue((uint32_t)(uintptr_t)&s_slab[(ks + 1) & 1][0][0],
                ws + OFF_W0B + (ks + 1) * SLAB_ELE);
    v16h a = load_a_frag(myArow, ks * 32, sel8);
    v16h b = *(const v16h*)(&s_slab[ks & 1][nr][0] + sel16);
    accf = __builtin_amdgcn_wmma_f32_16x16x32_f16(
        false, a, false, b, (short)0, accf, false, false);
  }
#else
#pragma unroll 1
  for (int ks = 0; ks < 8; ++ks) {
    __syncthreads();
    {
      const float4* src = (const float4*)(ws + OFF_W0B + ks * SLAB_ELE);
      float4* dst = (float4*)&s_slab[0][0][0];
#pragma unroll
      for (int q = 0; q < 4; ++q) dst[q * 256 + tid] = src[q * 256 + tid];
    }
    __syncthreads();
    v16h a = load_a_frag(myArow, ks * 32, sel8);
    v16h b = *(const v16h*)(&s_slab[0][nr][0] + sel16);
    accf = __builtin_amdgcn_wmma_f32_16x16x32_f16(
        false, a, false, b, (short)0, accf, false, false);
  }
#endif

  // C layout -> per-pair lanes via LDS
#pragma unroll
  for (int v = 0; v < 8; ++v) s_gf[wave][v + moff][nr] = accf[v];
  __syncthreads();

  if (lane < 16) {
    const float g0 = s_gf[wave][lane][0];
    const float g1 = s_gf[wave][lane][1];
    const float g2 = s_gf[wave][lane][2];
    const float g3 = s_gf[wave][lane][3];   // dE/dR feature
    const float g4 = s_gf[wave][lane][4];   // dE/d(1/R) feature
    const float Rinv = 1.0f / R;
    const float r2 = Rinv * Rinv;
    const float r4 = r2 * r2;
    const float r8 = r4 * r4;
    const float r14 = r8 * r4 * r2;
    // dR/ddr = dr/R ; d(1/R)/ddr = -dr/R^3 ; dU0/ddr = -12 * dr / R^14 (sigma=1, n=12)
    const float coef = g3 * Rinv - g4 * r2 * Rinv - 12.0f * r14;
    const float fx = -(g0 + coef * dx);
    const float fy = -(g1 + coef * dy);
    const float fz = -(g2 + coef * dz);
    const int arow = wave * 16 + lane;      // atom-in-block = arow/32
    atomicAdd(&s_fsum[arow >> 5][0], fx);
    atomicAdd(&s_fsum[arow >> 5][1], fy);
    atomicAdd(&s_fsum[arow >> 5][2], fz);
  }
  __syncthreads();
  if (tid < 12)
    out[(blockIdx.x * 4 + tid / 3) * 3 + (tid % 3)] = s_fsum[tid / 3][tid % 3];
}

extern "C" void kernel_launch(void* const* d_in, const int* in_sizes, int n_in,
                              void* d_out, int out_size, void* d_ws, size_t ws_size,
                              hipStream_t stream) {
  (void)in_sizes; (void)n_in; (void)out_size; (void)ws_size;
  const float* pos = (const float*)d_in[0];
  const int*   nl  = (const int*)d_in[1];
  const float* W0  = (const float*)d_in[2];
  const float* b0  = (const float*)d_in[3];
  const float* W1  = (const float*)d_in[4];
  const float* b1  = (const float*)d_in[5];
  const float* W2  = (const float*)d_in[6];
  const float* b2  = (const float*)d_in[7];
  const float* W3  = (const float*)d_in[8];
  float* out = (float*)d_out;
  _Float16* ws = (_Float16*)d_ws;

  prep_weights_kernel<<<(WS_TOTAL + 255) / 256, 256, 0, stream>>>(W0, W1, W2, ws);

  const int pairs = NBAT * NATM * NNB;          // 262144
  dim3 grid(pairs / ROWS_PER_BLOCK);            // 2048 blocks
  pairmlp_force_kernel<<<grid, 256, 0, stream>>>(pos, nl, ws, b0, b1, b2, W3, out);
}
